// AttLayer_50955492000290
// MI455X (gfx1250) — compile-verified, matrix operand order checked by reading, not run on hardware
//
#include <hip/hip_runtime.h>

// ---------------------------------------------------------------------------
// Graph TransformerConv layer for MI455X (gfx1250, wave32, WMMA).
//   Phase A: bf16 WMMA GEMMs  (fc+relu -> h ; h -> q,k,v,skip)
//   Phase B: edge attention   (dot, segment softmax via atomics, scatter-add)
// ---------------------------------------------------------------------------

typedef __bf16 bf16_t;
typedef __attribute__((ext_vector_type(16))) __bf16 v16bf;
typedef __attribute__((ext_vector_type(8)))  __bf16 v8bf;
typedef __attribute__((ext_vector_type(8)))  float  v8f;

#define D_IN_  256
#define D_H_   128
#define D_OUT_ 128

// ---- fragment loaders (per CDNA5 ISA 16-bit A/B 16x32 VGPR layout) --------
// lanes 0-15: row = lane,     K chunks {k0+0..7,  k0+16..23}
// lanes16-31: row = lane-16,  K chunks {k0+8..15, k0+24..31}
__device__ __forceinline__ v16bf frag_bf16(const bf16_t* __restrict__ base,
                                           int lda, int row, int k0, int lane) {
  const int kb = k0 + ((lane >> 4) << 3);
  const bf16_t* p = base + (size_t)row * lda + kb;
  v8bf lo = *(const v8bf*)(p);        // 16B load
  v8bf hi = *(const v8bf*)(p + 16);   // 16B load
  return __builtin_shufflevector(lo, hi, 0,1,2,3,4,5,6,7,8,9,10,11,12,13,14,15);
}

__device__ __forceinline__ v16bf frag_f32(const float* __restrict__ base,
                                          int lda, int row, int k0, int lane) {
  const int kb = k0 + ((lane >> 4) << 3);
  const float* p = base + (size_t)row * lda + kb;
  v16bf r;
#pragma unroll
  for (int i = 0; i < 8; ++i) {
    r[i]     = (bf16_t)p[i];
    r[i + 8] = (bf16_t)p[i + 16];
  }
  return r;
}

// ---- weight conversion (f32 -> bf16, tiny, once per launch) ---------------
__global__ void k_convert_weights(const float* __restrict__ Wfc,
                                  const float* __restrict__ Wq,
                                  const float* __restrict__ Wk,
                                  const float* __restrict__ Wv,
                                  const float* __restrict__ Ws,
                                  bf16_t* __restrict__ WfcB,
                                  bf16_t* __restrict__ WcatB) {
  const int i = blockIdx.x * blockDim.x + threadIdx.x;
  if (i < D_H_ * D_IN_) WfcB[i] = (bf16_t)Wfc[i];
  if (i < D_OUT_ * D_H_) {
    WcatB[i]                       = (bf16_t)Wq[i];
    WcatB[1 * D_OUT_ * D_H_ + i]   = (bf16_t)Wk[i];
    WcatB[2 * D_OUT_ * D_H_ + i]   = (bf16_t)Wv[i];
    WcatB[3 * D_OUT_ * D_H_ + i]   = (bf16_t)Ws[i];
  }
}

// ---- GEMM 1: H = relu(X[N,256] * WfcT + b), H stored bf16 -----------------
// 256 threads = 8 waves; wave w -> rows [m0, m0+16), all 128 output cols.
__global__ void __launch_bounds__(256)
k_gemm_fc_relu(const float* __restrict__ X, const bf16_t* __restrict__ Wb,
               const float* __restrict__ bias, bf16_t* __restrict__ H, int Nn) {
  const int lane = threadIdx.x & 31;
  const int wave = threadIdx.x >> 5;
  const int m0 = (blockIdx.x * 8 + wave) * 16;
  if (m0 >= Nn) return;
  int arow = m0 + (lane & 15);
  if (arow >= Nn) arow = Nn - 1;          // clamp tail loads
  const int brow = lane & 15;

  v8f acc[8] = {};
  for (int k0 = 0; k0 < D_IN_; k0 += 32) {
    v16bf a = frag_f32(X, D_IN_, arow, k0, lane);
#pragma unroll
    for (int t = 0; t < 8; ++t) {
      v16bf b = frag_bf16(Wb, D_IN_, t * 16 + brow, k0, lane);
      acc[t] = __builtin_amdgcn_wmma_f32_16x16x32_bf16(
          false, a, false, b, (short)0, acc[t], false, false);
    }
  }
  // C/D layout: vgpr r, lanes0-15 -> M=r, lanes16-31 -> M=r+8; N = lane&15
  const int chi = (lane >> 4) << 3;
  const int col = lane & 15;
#pragma unroll
  for (int t = 0; t < 8; ++t) {
    const int n = t * 16 + col;
    const float bv = bias[n];
#pragma unroll
    for (int r = 0; r < 8; ++r) {
      const int row = m0 + chi + r;
      if (row < Nn) {
        float v = acc[t][r] + bv;
        v = fmaxf(v, 0.0f);
        H[(size_t)row * D_H_ + n] = (bf16_t)v;
      }
    }
  }
}

// ---- GEMM 2: {q,k,v,skip} = H[N,128] * W_yT + b_y -------------------------
// blockIdx.y in 0..3 selects output matrix; skip (y==3) goes straight to out.
__global__ void __launch_bounds__(256)
k_gemm_qkvs(const bf16_t* __restrict__ H, const bf16_t* __restrict__ Wcat,
            const float* __restrict__ bq, const float* __restrict__ bk,
            const float* __restrict__ bv, const float* __restrict__ bs,
            float* __restrict__ Q, float* __restrict__ K,
            float* __restrict__ V, float* __restrict__ Out, int Nn) {
  const int lane = threadIdx.x & 31;
  const int wave = threadIdx.x >> 5;
  const int y = blockIdx.y;
  const int m0 = (blockIdx.x * 8 + wave) * 16;
  if (m0 >= Nn) return;
  int arow = m0 + (lane & 15);
  if (arow >= Nn) arow = Nn - 1;
  const int brow = lane & 15;

  const bf16_t* __restrict__ Wy = Wcat + (size_t)y * D_OUT_ * D_H_;
  const float* __restrict__ by = (y == 0) ? bq : (y == 1) ? bk : (y == 2) ? bv : bs;
  float* __restrict__ O = (y == 0) ? Q : (y == 1) ? K : (y == 2) ? V : Out;

  v8f acc[8] = {};
  for (int k0 = 0; k0 < D_H_; k0 += 32) {
    v16bf a = frag_bf16(H, D_H_, arow, k0, lane);
#pragma unroll
    for (int t = 0; t < 8; ++t) {
      v16bf b = frag_bf16(Wy, D_H_, t * 16 + brow, k0, lane);
      acc[t] = __builtin_amdgcn_wmma_f32_16x16x32_bf16(
          false, a, false, b, (short)0, acc[t], false, false);
    }
  }
  const int chi = (lane >> 4) << 3;
  const int col = lane & 15;
#pragma unroll
  for (int t = 0; t < 8; ++t) {
    const int n = t * 16 + col;
    const float bb = by[n];
#pragma unroll
    for (int r = 0; r < 8; ++r) {
      const int row = m0 + chi + r;
      if (row < Nn) O[(size_t)row * D_OUT_ + n] = acc[t][r] + bb;
    }
  }
}

// ---- order-preserving f32 <-> u32 encoding for atomic segment-max ---------
__device__ __forceinline__ unsigned enc_f32(float f) {
  unsigned u = __float_as_uint(f);
  return u ^ ((u >> 31) ? 0xFFFFFFFFu : 0x80000000u);
}
__device__ __forceinline__ float dec_f32(unsigned u) {
  unsigned v = (u >> 31) ? (u ^ 0x80000000u) : ~u;
  return __uint_as_float(v);
}

__global__ void k_init_nodes(unsigned* __restrict__ nmax,
                             float* __restrict__ denom, int Nn) {
  const int i = blockIdx.x * blockDim.x + threadIdx.x;
  if (i < Nn) { nmax[i] = 0u; denom[i] = 0.0f; }
}

// ---- edge pass 1: score = <q[dst],k[src]>/sqrt(D); atomic segment max -----
// one wave32 per edge, lane = float4 slice (coalesced 512B L2-resident gather)
__global__ void __launch_bounds__(256)
k_edge_scores(const float* __restrict__ Q, const float* __restrict__ K,
              const int* __restrict__ ei, float* __restrict__ scores,
              unsigned* __restrict__ nmax, int Ee) {
  const int e = (int)((blockIdx.x * blockDim.x + threadIdx.x) >> 5);
  const int lane = threadIdx.x & 31;
  if (e >= Ee) return;
  const int src = ei[e];
  const int dst = ei[Ee + e];
  const float4 qv = *(const float4*)(Q + (size_t)dst * D_OUT_ + lane * 4);
  const float4 kv = *(const float4*)(K + (size_t)src * D_OUT_ + lane * 4);
  float s = qv.x * kv.x + qv.y * kv.y + qv.z * kv.z + qv.w * kv.w;
#pragma unroll
  for (int off = 16; off > 0; off >>= 1) s += __shfl_down(s, off, 32);
  if (lane == 0) {
    s *= 0.08838834764831845f;  // 1/sqrt(128)
    scores[e] = s;
    atomicMax(nmax + dst, enc_f32(s));
  }
}

// ---- edge pass 2: ex = exp(s - max[dst]); atomic segment sum --------------
__global__ void __launch_bounds__(256)
k_edge_exp(const int* __restrict__ ei, float* __restrict__ scores,
           const unsigned* __restrict__ nmax, float* __restrict__ denom, int Ee) {
  const int e = blockIdx.x * blockDim.x + threadIdx.x;
  if (e >= Ee) return;
  const int dst = ei[Ee + e];
  const float ex = __expf(scores[e] - dec_f32(nmax[dst]));
  scores[e] = ex;              // reuse buffer for ex
  atomicAdd(denom + dst, ex);
}

// ---- edge pass 3: out[dst] += (ex/denom[dst]) * v[src] --------------------
__global__ void __launch_bounds__(256)
k_edge_agg(const float* __restrict__ V, const int* __restrict__ ei,
           const float* __restrict__ ex, const float* __restrict__ denom,
           float* __restrict__ Out, int Ee) {
  const int e = (int)((blockIdx.x * blockDim.x + threadIdx.x) >> 5);
  const int lane = threadIdx.x & 31;
  if (e >= Ee) return;
  const int src = ei[e];
  const int dst = ei[Ee + e];
  const float coef = ex[e] / denom[dst];
  const float4 vv = *(const float4*)(V + (size_t)src * D_OUT_ + lane * 4);
  float* o = Out + (size_t)dst * D_OUT_ + lane * 4;
  atomicAdd(o + 0, coef * vv.x);
  atomicAdd(o + 1, coef * vv.y);
  atomicAdd(o + 2, coef * vv.z);
  atomicAdd(o + 3, coef * vv.w);
}

// ---------------------------------------------------------------------------
extern "C" void kernel_launch(void* const* d_in, const int* in_sizes, int n_in,
                              void* d_out, int out_size, void* d_ws, size_t ws_size,
                              hipStream_t stream) {
  const float* x   = (const float*)d_in[0];
  const int*   ei  = (const int*)d_in[1];     // [2,E] int32 (jax default x64 off)
  const float* Wfc = (const float*)d_in[2];
  const float* bfc = (const float*)d_in[3];
  const float* Wq  = (const float*)d_in[4];
  const float* bq  = (const float*)d_in[5];
  const float* Wk  = (const float*)d_in[6];
  const float* bk  = (const float*)d_in[7];
  const float* Wv  = (const float*)d_in[8];
  const float* bv  = (const float*)d_in[9];
  const float* Ws  = (const float*)d_in[10];
  const float* bs  = (const float*)d_in[11];
  float* out = (float*)d_out;

  const int Nn = in_sizes[0] / D_IN_;   // 50000
  const int Ee = in_sizes[1] / 2;       // 800000

  // workspace carve-out (256B aligned)
  char* base = (char*)d_ws;
  size_t off = 0;
  auto carve = [&](size_t bytes) {
    char* p = base + off;
    off += (bytes + 255) & ~(size_t)255;
    return p;
  };
  bf16_t*  Hb    = (bf16_t*)carve((size_t)Nn * D_H_ * sizeof(bf16_t));
  bf16_t*  WfcB  = (bf16_t*)carve((size_t)D_H_ * D_IN_ * sizeof(bf16_t));
  bf16_t*  WcatB = (bf16_t*)carve((size_t)4 * D_OUT_ * D_H_ * sizeof(bf16_t));
  float*   Qf    = (float*)carve((size_t)Nn * D_OUT_ * sizeof(float));
  float*   Kf    = (float*)carve((size_t)Nn * D_OUT_ * sizeof(float));
  float*   Vf    = (float*)carve((size_t)Nn * D_OUT_ * sizeof(float));
  float*   sc    = (float*)carve((size_t)Ee * sizeof(float));
  unsigned* nmax = (unsigned*)carve((size_t)Nn * sizeof(unsigned));
  float*   denom = (float*)carve((size_t)Nn * sizeof(float));
  (void)ws_size; (void)n_in; (void)out_size;

  const int cvt_blocks = (D_H_ * D_IN_ + 255) / 256;
  k_convert_weights<<<cvt_blocks, 256, 0, stream>>>(Wfc, Wq, Wk, Wv, Ws, WfcB, WcatB);

  const int gemm_blocks = (Nn + 127) / 128;   // 8 waves * 16 rows per block
  k_gemm_fc_relu<<<gemm_blocks, 256, 0, stream>>>(x, WfcB, bfc, Hb, Nn);
  k_gemm_qkvs<<<dim3(gemm_blocks, 4), 256, 0, stream>>>(
      Hb, WcatB, bq, bk, bv, bs, Qf, Kf, Vf, out, Nn);

  k_init_nodes<<<(Nn + 255) / 256, 256, 0, stream>>>(nmax, denom, Nn);

  const int wave_blocks = (Ee + 7) / 8;       // 8 waves/block, 1 edge per wave
  k_edge_scores<<<wave_blocks, 256, 0, stream>>>(Qf, Kf, ei, sc, nmax, Ee);
  k_edge_exp<<<(Ee + 255) / 256, 256, 0, stream>>>(ei, sc, nmax, denom, Ee);
  k_edge_agg<<<wave_blocks, 256, 0, stream>>>(Vf, ei, sc, denom, out, Ee);
}
